// Sub2GeneDifferCrossAttn_42073499631975
// MI455X (gfx1250) — matrix operand order repaired and checked by reading, not compiled, not run on hardware
//
#include <hip/hip_runtime.h>
#include <cstdint>
#include <cstddef>

// ---------------------------------------------------------------------------
// Problem constants (from reference)
// ---------------------------------------------------------------------------
constexpr int Bv  = 16;
constexpr int Sv  = 1024;
constexpr int Ev  = 512;
constexpr int Lv  = 2048;          // P*G
constexpr int HDv = 256;           // head dim
constexpr float SCALEv      = 0.0625f;   // HD^-0.5 = 1/16
constexpr float LAMBDA_INIT = 0.2f;      // 0.8 - 0.6*exp(0)
constexpr float ONE_MINUS_L = 0.8f;
constexpr float EPSv        = 1e-5f;

typedef __attribute__((ext_vector_type(16))) __bf16 v16bf;
typedef __attribute__((ext_vector_type(8)))  float  v8f;

// ---------------------------------------------------------------------------
// CDNA5 async global<->LDS copies (ASYNCcnt path, CDNA5 ISA)
// ---------------------------------------------------------------------------
__device__ __forceinline__ unsigned lds_addr_of(const void* p) {
  // generic->LDS: low 32 bits of a flat shared pointer are the LDS byte offset
  return (unsigned)(uintptr_t)p;
}
__device__ __forceinline__ void async_copy_b128(unsigned ldsAddr, unsigned gOff,
                                                unsigned long long sbase) {
  asm volatile("global_load_async_to_lds_b128 %0, %1, %2 offset:0"
               :: "v"(ldsAddr), "v"(gOff), "s"(sbase)
               : "memory");
}
__device__ __forceinline__ void async_store_b128(unsigned gOff, unsigned ldsAddr,
                                                 unsigned long long sbase) {
  asm volatile("global_store_async_from_lds_b128 %0, %1, %2 offset:0"
               :: "v"(gOff), "v"(ldsAddr), "s"(sbase)
               : "memory");
}
template <int N>
__device__ __forceinline__ void wait_async_le() {
  asm volatile("s_wait_asynccnt %0" :: "i"(N) : "memory");
}

// ---------------------------------------------------------------------------
// WMMA helpers (CDNA5 wave32, v_wmma_f32_16x16x32_bf16)
// ---------------------------------------------------------------------------
__device__ __forceinline__ v8f wmma_bf16(v16bf a, v16bf b, v8f c) {
  return __builtin_amdgcn_wmma_f32_16x16x32_bf16(
      /*neg_a=*/false, a, /*neg_b=*/false, b,
      /*c_mod=*/(short)0, c, /*reuse_a=*/false, /*reuse_b=*/false);
}

// A-matrix fragment 16x32 bf16 from row-major [16][ld] LDS region.
__device__ __forceinline__ v16bf load_a_k32(const __bf16* base, int ld) {
  const int lane = threadIdx.x & 31;
  const int m  = lane & 15;
  const int hi = lane >> 4;
  const unsigned int* p = reinterpret_cast<const unsigned int*>(base + m * ld);
  union { unsigned int u[8]; v16bf v; } f;
  const int k0 = hi * 4;
  const int k1 = 8 + hi * 4;
#pragma unroll
  for (int i = 0; i < 4; ++i) { f.u[i] = p[k0 + i]; f.u[4 + i] = p[k1 + i]; }
  return f.v;
}

// B-matrix fragment 32x16 bf16; source stored N-major: [16][ld], K contiguous.
__device__ __forceinline__ v16bf load_b_k32(const __bf16* base, int ld) {
  const int lane = threadIdx.x & 31;
  const int n  = lane & 15;
  const int hi = lane >> 4;
  const unsigned int* p = reinterpret_cast<const unsigned int*>(base + n * ld);
  union { unsigned int u[8]; v16bf v; } f;
#pragma unroll
  for (int i = 0; i < 4; ++i) { f.u[i] = p[hi * 4 + i]; f.u[4 + i] = p[8 + hi * 4 + i]; }
  return f.v;
}

// ---------------------------------------------------------------------------
// Kernel 1: C_bf16[M,512] = (A_f32[M,512] @ W_f32[512,512]) * alpha
// Epilogue: C tile via LDS + global_store_async_from_lds_b128.
// ---------------------------------------------------------------------------
__global__ __launch_bounds__(256) void proj_gemm_f32A(
    const float* __restrict__ A, const float* __restrict__ W,
    __bf16* __restrict__ Cb, float alpha) {
  __shared__ __bf16 sA[128 * 32];
  __shared__ __bf16 sWt[64 * 32];
  __shared__ __bf16 sC[128 * 64];
  const int tid = threadIdx.x;
  const int lane = tid & 31, w = tid >> 5;
  const int wm = w >> 1, wn = w & 1;
  const int mblk = blockIdx.x * 128;
  const int nblk = blockIdx.y * 64;
  v8f acc[2][2] = {};

  for (int kc = 0; kc < Ev; kc += 32) {
    __syncthreads();
#pragma unroll
    for (int i = 0; i < 4; ++i) {          // A tile 128x32 f32->bf16
      int e = tid + i * 256;
      int r = e >> 3, q = e & 7;
      union { float4 f4; float f[4]; } v;
      v.f4 = *(const float4*)&A[(size_t)(mblk + r) * Ev + kc + q * 4];
      union { __bf16 h[4]; uint2 u; } o;
#pragma unroll
      for (int j = 0; j < 4; ++j) o.h[j] = (__bf16)v.f[j];
      *(uint2*)&sA[r * 32 + q * 4] = o.u;
    }
#pragma unroll
    for (int i = 0; i < 2; ++i) {          // W tile 32x64 transposed
      int e = tid + i * 256;
      int k = e >> 4, nq = e & 15;
      union { float4 f4; float f[4]; } v;
      v.f4 = *(const float4*)&W[(size_t)(kc + k) * Ev + nblk + nq * 4];
#pragma unroll
      for (int j = 0; j < 4; ++j) sWt[(nq * 4 + j) * 32 + k] = (__bf16)v.f[j];
    }
    __syncthreads();
#pragma unroll
    for (int ms = 0; ms < 2; ++ms) {
      v16bf a = load_a_k32(&sA[(wm * 32 + ms * 16) * 32], 32);
#pragma unroll
      for (int ns = 0; ns < 2; ++ns) {
        v16bf b = load_b_k32(&sWt[(wn * 32 + ns * 16) * 32], 32);
        acc[ms][ns] = wmma_bf16(a, b, acc[ms][ns]);
      }
    }
  }
  // epilogue: C tile -> LDS -> async b128 coalesced stores
  const int cm = (lane >> 4) * 8, cn = lane & 15;
#pragma unroll
  for (int ms = 0; ms < 2; ++ms)
#pragma unroll
    for (int ns = 0; ns < 2; ++ns) {
      union { v8f v; float f[8]; } u; u.v = acc[ms][ns];
#pragma unroll
      for (int v = 0; v < 8; ++v)
        sC[(wm * 32 + ms * 16 + v + cm) * 64 + wn * 32 + ns * 16 + cn] =
            (__bf16)(u.f[v] * alpha);
    }
  __syncthreads();
  {
    const unsigned sClds = lds_addr_of(sC);
    unsigned long long sbase =
        (unsigned long long)(const void*)(Cb + (size_t)mblk * Ev + nblk);
#pragma unroll
    for (int i = 0; i < 4; ++i) {          // 16 KB tile, 128B rows @1024B stride
      unsigned beta = (unsigned)(tid * 16 + i * 4096);
      unsigned row = beta >> 7, inrow = beta & 127;
      async_store_b128(row * (Ev * 2) + inrow, sClds + beta, sbase);
    }
  }  // S_ENDPGM waits all counters
}

// ---------------------------------------------------------------------------
// Kernel 3: out_f32[M,512] = Rb_bf16[M,512] @ Wo_f32[512,512]
// A staged by async load; C drained by async store.
// ---------------------------------------------------------------------------
__global__ __launch_bounds__(256) void final_gemm_bf16A(
    const __bf16* __restrict__ A, const float* __restrict__ W,
    float* __restrict__ C) {
  __shared__ __bf16 sA[128 * 32];
  __shared__ __bf16 sWt[64 * 32];
  __shared__ float  sC[128 * 64];
  const int tid = threadIdx.x;
  const int lane = tid & 31, w = tid >> 5;
  const int wm = w >> 1, wn = w & 1;
  const int mblk = blockIdx.x * 128;
  const int nblk = blockIdx.y * 64;
  const unsigned sAlds = lds_addr_of(&sA[0]);
  v8f acc[2][2] = {};

  for (int kc = 0; kc < Ev; kc += 32) {
    __syncthreads();
    {
      unsigned long long sbase =
          (unsigned long long)(const void*)(A + (size_t)mblk * Ev + kc);
#pragma unroll
      for (int i = 0; i < 2; ++i) {        // 8 KB A tile, 64B rows @1024B stride
        unsigned beta = (unsigned)(tid * 16 + i * 4096);
        unsigned row = beta >> 6, inrow = beta & 63;
        async_copy_b128(sAlds + beta, row * (Ev * 2) + inrow, sbase);
      }
    }
#pragma unroll
    for (int i = 0; i < 2; ++i) {          // W tile, overlaps async
      int e = tid + i * 256;
      int k = e >> 4, nq = e & 15;
      union { float4 f4; float f[4]; } v;
      v.f4 = *(const float4*)&W[(size_t)(kc + k) * Ev + nblk + nq * 4];
#pragma unroll
      for (int j = 0; j < 4; ++j) sWt[(nq * 4 + j) * 32 + k] = (__bf16)v.f[j];
    }
    wait_async_le<0>();
    __syncthreads();
#pragma unroll
    for (int ms = 0; ms < 2; ++ms) {
      v16bf a = load_a_k32(&sA[(wm * 32 + ms * 16) * 32], 32);
#pragma unroll
      for (int ns = 0; ns < 2; ++ns) {
        v16bf b = load_b_k32(&sWt[(wn * 32 + ns * 16) * 32], 32);
        acc[ms][ns] = wmma_bf16(a, b, acc[ms][ns]);
      }
    }
  }
  const int cm = (lane >> 4) * 8, cn = lane & 15;
#pragma unroll
  for (int ms = 0; ms < 2; ++ms)
#pragma unroll
    for (int ns = 0; ns < 2; ++ns) {
      union { v8f v; float f[8]; } u; u.v = acc[ms][ns];
#pragma unroll
      for (int v = 0; v < 8; ++v)
        sC[(wm * 32 + ms * 16 + v + cm) * 64 + wn * 32 + ns * 16 + cn] = u.f[v];
    }
  __syncthreads();
  {
    const unsigned sClds = lds_addr_of(sC);
    unsigned long long sbase =
        (unsigned long long)(const void*)(C + (size_t)mblk * Ev + nblk);
#pragma unroll
    for (int i = 0; i < 8; ++i) {          // 32 KB tile, 256B rows @2048B stride
      unsigned beta = (unsigned)(tid * 16 + i * 4096);
      unsigned row = beta >> 8, inrow = beta & 255;
      async_store_b128(row * (Ev * 4) + inrow, sClds + beta, sbase);
    }
  }
}

// ---------------------------------------------------------------------------
// Kernel 2: fused dual-head attention per (batch, 16 query rows).
//   K staged by double-buffered async b128 (partial s_wait_asynccnt 8),
//   scores WMMA -> masked softmax -> diff (async-drained from LDS while
//   diff@V WMMA runs) -> RMSNorm epilogue -> Rb via async store.
// Dynamic LDS layout (bytes):
//   sQ    16*512  bf16   @0        (16384)
//   stage 32768   bf16   @16384    (65536)  K halves 2x32KB | V^T | Rb tile
//   sS    16*2048 f32    @81920    (131072)
//   sD    16*2048 bf16   @212992   (65536)
//   red   256 f32        @278528 ; rowm/rows/rowss 16 f32 each ; lam f32
// ---------------------------------------------------------------------------
constexpr size_t SMEM_BYTES = 279808;

__global__ __launch_bounds__(256) void fused_attn(
    const __bf16* __restrict__ Qb, const __bf16* __restrict__ Kb,
    const __bf16* __restrict__ Vb, const uint8_t* __restrict__ mask,
    const float* __restrict__ lq1, const float* __restrict__ lk1,
    const float* __restrict__ lq2, const float* __restrict__ lk2,
    const float* __restrict__ gamma,
    float* __restrict__ outDiff, __bf16* __restrict__ Rb) {
  extern __shared__ char sm[];
  __bf16* sQ     = (__bf16*)sm;
  __bf16* sStage = (__bf16*)(sm + 16384);
  float*  sS     = (float*)(sm + 81920);
  __bf16* sD     = (__bf16*)(sm + 212992);
  float*  red    = (float*)(sm + 278528);
  float*  rowm   = red + 256;
  float*  rowsum = rowm + 16;
  float*  rowss  = rowsum + 16;
  float*  sLam   = rowss + 16;

  const int tid  = threadIdx.x;
  const int lane = tid & 31;
  const int w    = tid >> 5;
  const int s0   = blockIdx.x * 16;
  const int b    = blockIdx.y;
  const unsigned sQlds = lds_addr_of(sQ);
  const unsigned stageLds = lds_addr_of(sStage);

  // ---- kick off async Q tile copy: 16 rows x 1024B, contiguous 16 KB
  {
    unsigned long long sbase =
        (unsigned long long)(const void*)(Qb + (size_t)(b * Sv + s0) * Ev);
#pragma unroll
    for (int i = 0; i < 4; ++i) {
      unsigned beta = (unsigned)(tid * 16 + i * 4096);
      async_copy_b128(sQlds + beta, beta, sbase);
    }
  }

  // ---- lambda (overlaps async Q)
  {
    float* red2 = sS;
    red[tid]  = lq1[tid] * lk1[tid];
    red2[tid] = lq2[tid] * lk2[tid];
    __syncthreads();
    for (int o = 128; o > 0; o >>= 1) {
      if (tid < o) { red[tid] += red[tid + o]; red2[tid] += red2[tid + o]; }
      __syncthreads();
    }
    if (tid == 0) *sLam = __expf(red[0]) - __expf(red2[0]) + LAMBDA_INIT;
    __syncthreads();
  }
  const float lam = *sLam;
  wait_async_le<0>();
  __syncthreads();

  // ---- scores + softmax per head (double-buffered K staging)
  for (int h = 0; h < 2; ++h) {
    // issue chunk 0 into half 0 (8 async b128 per thread = 32 KB / WG)
    {
      unsigned long long sbase = (unsigned long long)(const void*)
          (Kb + ((size_t)b * Lv + 0 * 128) * Ev + h * HDv);
#pragma unroll
      for (int i = 0; i < 8; ++i) {
        unsigned beta = (unsigned)(tid * 16 + i * 4096);
        unsigned row = beta >> 9, inrow = beta & 511;
        async_copy_b128(stageLds + beta, row * (Ev * 2) + inrow, sbase);
      }
    }
    for (int c = 0; c < 16; ++c) {
      if (c < 15) {  // prefetch next chunk into the other half
        unsigned long long sbase = (unsigned long long)(const void*)
            (Kb + ((size_t)b * Lv + (c + 1) * 128) * Ev + h * HDv);
        unsigned half = (unsigned)((c + 1) & 1) * 32768u;
#pragma unroll
        for (int i = 0; i < 8; ++i) {
          unsigned beta = (unsigned)(tid * 16 + i * 4096);
          unsigned row = beta >> 9, inrow = beta & 511;
          async_copy_b128(stageLds + half + beta, row * (Ev * 2) + inrow, sbase);
        }
        wait_async_le<8>();   // chunk c resident; c+1 still in flight
      } else {
        wait_async_le<0>();
      }
      __syncthreads();
      const __bf16* kHalf = sStage + (size_t)(c & 1) * 16384;
      v8f acc = {};
#pragma unroll
      for (int kk = 0; kk < 8; ++kk) {               // HD=256 = 8 * 32
        v16bf a = load_a_k32(sQ + h * HDv + kk * 32, Ev);
        v16bf bb = load_b_k32(kHalf + (w * 16) * HDv + kk * 32, HDv);
        acc = wmma_bf16(a, bb, acc);
      }
      union { v8f v; float f[8]; } u; u.v = acc;
      const int cm = (lane >> 4) * 8, cn = lane & 15;
#pragma unroll
      for (int v = 0; v < 8; ++v)
        sS[(v + cm) * Lv + c * 128 + w * 16 + cn] = u.f[v];
      __syncthreads();        // protect this half before it is re-staged
    }

    // masked softmax over sS rows; 16 threads per row, 128 elems each
    {
      const int r = tid >> 4, seg = tid & 15;
      float* row = sS + r * Lv + seg * 128;
      if (!mask[b * Sv + s0 + r]) {
        for (int i = 0; i < 128; ++i) row[i] = 0.f;
      }
      float mx = -3.4e38f;
      for (int i = 0; i < 128; ++i) mx = fmaxf(mx, row[i]);
      red[tid] = mx;
      __syncthreads();
      if (seg == 0) {
        float m2 = red[tid];
        for (int j = 1; j < 16; ++j) m2 = fmaxf(m2, red[tid + j]);
        rowm[r] = m2;
      }
      __syncthreads();
      const float rm = rowm[r];
      float ssum = 0.f;
      for (int i = 0; i < 128; ++i) {
        float e = __expf(row[i] - rm); row[i] = e; ssum += e;
      }
      red[tid] = ssum;
      __syncthreads();
      if (seg == 0) {
        float s2 = 0.f;
        for (int j = 0; j < 16; ++j) s2 += red[tid + j];
        rowsum[r] = s2;
      }
      __syncthreads();
      const float inv = 1.f / rowsum[r];
      __bf16* drow = sD + r * Lv + seg * 128;
      if (h == 0) {
        for (int i = 0; i < 128; ++i) drow[i] = (__bf16)(row[i] * inv);
      } else {
        for (int i = 0; i < 128; ++i) {
          float dv = (float)drow[i] - lam * (row[i] * inv);
          row[i]  = dv;                 // diff kept in sS, drained async below
          drow[i] = (__bf16)dv;
        }
      }
    }
    __syncthreads();
  }

  // ---- drain diff (16x2048 f32 = 128 KB, contiguous in global) from LDS,
  //      overlapping the diff@V WMMA phase below
  {
    unsigned long long sbase = (unsigned long long)(const void*)
        (outDiff + (size_t)(b * Sv + s0) * Lv);
    const unsigned sSlds = lds_addr_of(sS);
#pragma unroll
    for (int i = 0; i < 32; ++i) {
      unsigned beta = (unsigned)(tid * 16 + i * 4096);
      async_store_b128(beta, sSlds + beta, sbase);
    }
  }

  // ---- out2d[16,512] = diff[16,2048] @ V[2048,512] (V staged transposed)
  v8f accO[4] = {};
  const int nb = w * 64;
  for (int c = 0; c < 32; ++c) {
    __syncthreads();
    const __bf16* vbase = Vb + ((size_t)b * Lv + c * 64) * Ev;
#pragma unroll
    for (int i = 0; i < 16; ++i) {                    // 4096 uint4 / 256 thr
      int e = tid + i * 256;
      int rrow = e >> 6, q = e & 63;
      union { uint4 u; __bf16 hh[8]; } v;
      v.u = *(const uint4*)(vbase + (size_t)rrow * Ev + q * 8);
#pragma unroll
      for (int j = 0; j < 8; ++j) sStage[(q * 8 + j) * 64 + rrow] = v.hh[j];
    }
    __syncthreads();
#pragma unroll
    for (int kk = 0; kk < 2; ++kk) {
      v16bf a = load_a_k32(sD + c * 64 + kk * 32, Lv);
#pragma unroll
      for (int j = 0; j < 4; ++j) {
        v16bf bb = load_b_k32(sStage + (nb + j * 16) * 64 + kk * 32, 64);
        accO[j] = wmma_bf16(a, bb, accO[j]);
      }
    }
  }
  __syncthreads();

  // ---- fused RMSNorm epilogue -> Rb tile in LDS -> async b128 store
  if (tid < 16) rowss[tid] = 0.f;
  __syncthreads();
  const int cm = (lane >> 4) * 8, cn = lane & 15;
  union { v8f v; float f[8]; } uo[4];
#pragma unroll
  for (int j = 0; j < 4; ++j) {
    uo[j].v = accO[j];
#pragma unroll
    for (int v = 0; v < 8; ++v)
      atomicAdd(&rowss[v + cm], uo[j].f[v] * uo[j].f[v]);   // ds_add_f32
  }
  __syncthreads();
#pragma unroll
  for (int j = 0; j < 4; ++j) {
#pragma unroll
    for (int v = 0; v < 8; ++v) {
      const int r = v + cm;
      const int col = nb + j * 16 + cn;
      const float scale =
          rsqrtf(rowss[r] * (1.f / (float)Ev) + EPSv) * ONE_MINUS_L;
      sStage[r * Ev + col] = (__bf16)(uo[j].f[v] * scale * gamma[col]);
    }
  }
  __syncthreads();
  {
    unsigned long long sbase = (unsigned long long)(const void*)
        (Rb + (size_t)(b * Sv + s0) * Ev);
#pragma unroll
    for (int i = 0; i < 4; ++i) {          // 16 KB contiguous
      unsigned beta = (unsigned)(tid * 16 + i * 4096);
      async_store_b128(beta, stageLds + beta, sbase);
    }
  }  // S_ENDPGM waits all counters
}

// ---------------------------------------------------------------------------
// Host launcher.  Workspace (bytes): Qb 16 MB | Kb 32 MB | Vb 32 MB | Rb 16 MB
// d_out: out f32 [B*S*E] (8,388,608) then diff f32 [B*S*L] (33,554,432)
// ---------------------------------------------------------------------------
extern "C" void kernel_launch(void* const* d_in, const int* in_sizes, int n_in,
                              void* d_out, int out_size, void* d_ws,
                              size_t ws_size, hipStream_t stream) {
  const float*   query = (const float*)d_in[0];
  const float*   key   = (const float*)d_in[1];
  const uint8_t* mask  = (const uint8_t*)d_in[2];
  const float*   Wq    = (const float*)d_in[3];
  const float*   Wk    = (const float*)d_in[4];
  const float*   Wv    = (const float*)d_in[5];
  const float*   Wo    = (const float*)d_in[6];
  const float*   lq1   = (const float*)d_in[7];
  const float*   lk1   = (const float*)d_in[8];
  const float*   lq2   = (const float*)d_in[9];
  const float*   lk2   = (const float*)d_in[10];
  const float*   gamma = (const float*)d_in[11];

  char* ws = (char*)d_ws;
  __bf16* Qb = (__bf16*)ws;                      ws += (size_t)Bv * Sv * Ev * 2;
  __bf16* Kb = (__bf16*)ws;                      ws += (size_t)Bv * Lv * Ev * 2;
  __bf16* Vb = (__bf16*)ws;                      ws += (size_t)Bv * Lv * Ev * 2;
  __bf16* Rb = (__bf16*)ws;

  float* out     = (float*)d_out;                          // [B*S, E]
  float* outDiff = (float*)d_out + (size_t)Bv * Sv * Ev;   // [B*S, L]

  // 1) projections
  {
    dim3 gq(Bv * Sv / 128, Ev / 64);
    dim3 gk(Bv * Lv / 128, Ev / 64);
    proj_gemm_f32A<<<gq, 256, 0, stream>>>(query, Wq, Qb, SCALEv);
    proj_gemm_f32A<<<gk, 256, 0, stream>>>(key,   Wk, Kb, 1.0f);
    proj_gemm_f32A<<<gk, 256, 0, stream>>>(key,   Wv, Vb, 1.0f);
  }
  // 2) fused attention + diff + RMSNorm
  {
    dim3 ga(Sv / 16, Bv);
    fused_attn<<<ga, 256, SMEM_BYTES, stream>>>(
        Qb, Kb, Vb, mask, lq1, lk1, lq2, lk2, gamma, outDiff, Rb);
  }
  // 3) output projection
  {
    dim3 go(Bv * Sv / 128, Ev / 64);
    final_gemm_bf16A<<<go, 256, 0, stream>>>(Rb, Wo, out);
  }
}
